// BucketAttentionLayer_41420664603301
// MI455X (gfx1250) — matrix operand order repaired and checked by reading, not compiled, not run on hardware
//
#include <hip/hip_runtime.h>

// ---------------------------------------------------------------------------
// BucketAttentionLayer for MI455X (gfx1250, wave32, WMMA)
// B=4, S=4096, D=1024, H=16, NB=32  ->  bucket=128 tokens, head_dim=64
// Strategy: bf16 WMMA (v_wmma_f32_16x16x32_bf16) for QKV projections and for
// both attention matmuls; f32 accumulate; f32 residual + output.
// LDS tiles staged with gfx1250 async loads (global_load_async_to_lds_b128,
// s_wait_asynccnt) when the toolchain exposes the builtins; otherwise the
// classic global_load_b128 + ds_store_b128 path.
// ---------------------------------------------------------------------------

static constexpr int Bb   = 4;
static constexpr int Sseq = 4096;
static constexpr int Dm   = 1024;
static constexpr int Hh   = 16;
static constexpr int NBkt = 32;
static constexpr int BSz  = 128;   // bucket size = Sseq / NBkt
static constexpr int HDm  = 64;    // head dim    = Dm / Hh

typedef __attribute__((ext_vector_type(16))) __bf16 v16bf;
typedef __attribute__((ext_vector_type(8)))  float  v8f;
typedef __attribute__((ext_vector_type(4)))  __bf16 v4bf;

union Frag16 { uint4 u[2]; v16bf v; };
union Pack8  { uint4 u; __bf16 h[8]; };

// ---- gfx1250 async LDS staging (guarded; falls back to VGPR round-trip) ----
#if defined(__HIP_DEVICE_COMPILE__) && defined(__gfx1250__) && \
    __has_builtin(__builtin_amdgcn_global_load_async_to_lds_b128)
#define USE_ASYNC_LDS 1
#else
#define USE_ASYNC_LDS 0
#endif

#if USE_ASYNC_LDS
// Builtin signature (per hipcc diagnostic): first param is a pointer to
// 'int __attribute__((vector_size(16)))' -- i.e. (v4i AS1*, v4i AS3*, Ii, Ii).
typedef int v4i32 __attribute__((vector_size(4 * sizeof(int))));
typedef v4i32 __attribute__((address_space(1)))* gptr_b128;
typedef v4i32 __attribute__((address_space(3)))* lptr_b128;

__device__ __forceinline__ void async_copy_b128(const void* g, void* l) {
  __builtin_amdgcn_global_load_async_to_lds_b128(
      (gptr_b128)(g), (lptr_b128)(l), 0, 0);
}
__device__ __forceinline__ void wait_async_zero() {
#if __has_builtin(__builtin_amdgcn_s_wait_asynccnt)
  __builtin_amdgcn_s_wait_asynccnt(0);
#else
  asm volatile("s_wait_asynccnt 0" ::: "memory");
#endif
}
#else
__device__ __forceinline__ void wait_async_zero() {}
#endif

__device__ __forceinline__ v8f wmma_bf16(v16bf a, v16bf b, v8f c) {
  // 8 args: (neg_a, A, neg_b, B, c_mod, C, reuse_a, reuse_b)
  return __builtin_amdgcn_wmma_f32_16x16x32_bf16(false, a, false, b,
                                                 (short)0, c, false, false);
}

// A-matrix fragment, 16(M) x 32(K) bf16, from a row-major LDS tile.
// ISA 7.12.2 layout: lane m=l&15, lanes16-31 shift K by +8;
// regs 0-3 hold K=ko..ko+7, regs 4-7 hold K=ko+16..ko+23.
__device__ __forceinline__ v16bf load_a_frag(const __bf16* base, int row0,
                                             int stride, int kbase) {
  const int lane = threadIdx.x & 31;
  const int m    = lane & 15;
  const int ko   = (lane >> 4) * 8;
  const __bf16* p = base + (size_t)(row0 + m) * stride + kbase + ko;
  Frag16 f;
  f.u[0] = *(const uint4*)(p);        // K = ko .. ko+7
  f.u[1] = *(const uint4*)(p + 16);   // K = ko+16 .. ko+23
  return f.v;
}

// B-matrix fragment, 32(K) x 16(N) bf16. Tile stored in LDS as [n][k] row-
// major (i.e. already B^T), so each lane n=l&15 reads K in the same order as A;
// lanes 16-31 shift K by +8 (regs 0-3: K=kh..kh+7, regs 4-7: K=kh+16..kh+23).
__device__ __forceinline__ v16bf load_b_frag(const __bf16* base, int col0,
                                             int stride, int kbase) {
  const int lane = threadIdx.x & 31;
  const int n    = lane & 15;
  const int kh   = (lane >> 4) * 8;
  const __bf16* p = base + (size_t)(col0 + n) * stride + kbase + kh;
  Frag16 f;
  f.u[0] = *(const uint4*)(p);        // K = kh .. kh+7
  f.u[1] = *(const uint4*)(p + 16);   // K = kh+16 .. kh+23
  return f.v;
}

// ---------------------------------------------------------------------------
// Kernel 0: f32 -> bf16 conversion (x and weights), 4 elems / thread.
// ---------------------------------------------------------------------------
__global__ void __launch_bounds__(256)
f32_to_bf16_kernel(const float* __restrict__ src, __bf16* __restrict__ dst,
                   int n4) {
  int i = blockIdx.x * blockDim.x + threadIdx.x;
  if (i < n4) {
    float4 f = ((const float4*)src)[i];
    v4bf o;
    o.x = (__bf16)f.x; o.y = (__bf16)f.y; o.z = (__bf16)f.z; o.w = (__bf16)f.w;
    ((v4bf*)dst)[i] = o;
  }
}

// ---------------------------------------------------------------------------
// Kernel 1: y = x @ W^T + b  (M=16384, N=1024, K=1024), bf16 in/out, f32 acc.
// Workgroup tile 128x128, k-slice 32. 8 waves: wave = (mq 0..3) x (nh 0..1),
// each wave produces 32(M) x 64(N) = 2x4 WMMA tiles.
// ---------------------------------------------------------------------------
__global__ void __launch_bounds__(256)
qkv_gemm_kernel(const __bf16* __restrict__ xb, const __bf16* __restrict__ wb,
                const float* __restrict__ bias, __bf16* __restrict__ yb) {
  __shared__ __align__(16) __bf16 smA[128 * 32];   // x rows  [m][k]
  __shared__ __align__(16) __bf16 smB[128 * 32];   // W rows  [n][k] == B^T

  const int tid  = threadIdx.x;
  const int lane = tid & 31;
  const int wv   = tid >> 5;
  const int mq   = wv >> 1;   // 32-row band within tile
  const int nh   = wv & 1;    // 64-col band within tile

  const int nBase = blockIdx.x * 128;
  const int mBase = blockIdx.y * 128;

  const v8f vzero = {0.f, 0.f, 0.f, 0.f, 0.f, 0.f, 0.f, 0.f};
  v8f acc[2][4];
#pragma unroll
  for (int mt = 0; mt < 2; ++mt)
#pragma unroll
    for (int nt = 0; nt < 4; ++nt) acc[mt][nt] = vzero;

  for (int kb = 0; kb < Dm; kb += 32) {
    __syncthreads();   // previous iteration's fragment reads complete
#pragma unroll
    for (int cc = 0; cc < 2; ++cc) {
      int c = tid + 256 * cc;              // 512 chunks of 8 bf16 per tile
      int row = c >> 2, off = (c & 3) * 8;
#if USE_ASYNC_LDS
      async_copy_b128(&xb[(size_t)(mBase + row) * Dm + kb + off],
                      &smA[row * 32 + off]);
      async_copy_b128(&wb[(size_t)(nBase + row) * Dm + kb + off],
                      &smB[row * 32 + off]);
#else
      *(uint4*)&smA[row * 32 + off] =
          *(const uint4*)&xb[(size_t)(mBase + row) * Dm + kb + off];
      *(uint4*)&smB[row * 32 + off] =
          *(const uint4*)&wb[(size_t)(nBase + row) * Dm + kb + off];
#endif
    }
    wait_async_zero();
    __syncthreads();

    if (kb + 32 < Dm) {   // pull next k-slice lines toward L2 (global_prefetch_b8)
      __builtin_prefetch(&xb[(size_t)(mBase + (tid >> 1)) * Dm + kb + 32], 0, 1);
      __builtin_prefetch(&wb[(size_t)(nBase + (tid >> 1)) * Dm + kb + 32], 0, 1);
    }

    v16bf a0 = load_a_frag(smA, 32 * mq, 32, 0);
    v16bf a1 = load_a_frag(smA, 32 * mq + 16, 32, 0);
#pragma unroll
    for (int nt = 0; nt < 4; ++nt) {
      v16bf bf_ = load_b_frag(smB, 64 * nh + 16 * nt, 32, 0);
      acc[0][nt] = wmma_bf16(a0, bf_, acc[0][nt]);
      acc[1][nt] = wmma_bf16(a1, bf_, acc[1][nt]);
    }
  }

  // Epilogue: C layout -> lane l, reg r: row = r + 8*(l>>4), col = l&15.
#pragma unroll
  for (int mt = 0; mt < 2; ++mt)
#pragma unroll
    for (int nt = 0; nt < 4; ++nt) {
      int col   = nBase + 64 * nh + 16 * nt + (lane & 15);
      float bvs = bias[col];
#pragma unroll
      for (int r = 0; r < 8; ++r) {
        int row = mBase + 32 * mq + 16 * mt + r + 8 * (lane >> 4);
        yb[(size_t)row * Dm + col] = (__bf16)(acc[mt][nt][r] + bvs);
      }
    }
}

// ---------------------------------------------------------------------------
// Kernel 2: bucket attention. One workgroup per (b, bucket, head).
// 8 waves; wave w owns score rows [16w, 16w+16). Scores live in registers
// (8 x v8f per wave = full 16x128 row block); softmax via 16-lane shfl_xor
// reductions; probs -> LDS (reusing Q/K region) as bf16 A-operand; P@V WMMA;
// fused residual add, f32 store.
// ---------------------------------------------------------------------------
__global__ void __launch_bounds__(256)
bucket_attn_kernel(const __bf16* __restrict__ qb, const __bf16* __restrict__ kb,
                   const __bf16* __restrict__ vb, const float* __restrict__ x,
                   float* __restrict__ out) {
  __shared__ __align__(16) __bf16 smQK[BSz * HDm * 2]; // Q[128][64] | K[128][64]; reused as P[128][128]
  __shared__ __align__(16) __bf16 smVt[HDm * BSz];     // V^T [64][128]

  const int tid  = threadIdx.x;
  const int lane = tid & 31;
  const int wv   = tid >> 5;

  const int bid = blockIdx.x;
  const int h   = bid & (Hh - 1);
  const int nb  = (bid >> 4) & (NBkt - 1);
  const int b   = bid >> 9;
  const int s0  = nb * BSz;

  __bf16* Qs = smQK;
  __bf16* Ks = smQK + BSz * HDm;
  __bf16* Ps = smQK;           // overlays Q+K after scores are in registers

  // Stage Q, K row-major (async when available) and V transposed into LDS.
#pragma unroll
  for (int cc = 0; cc < 4; ++cc) {
    int c = tid + 256 * cc;                 // 1024 chunks of 8 bf16 per tile
    int row = c >> 3, off = (c & 7) * 8;
    size_t g = ((size_t)(b * Sseq + s0 + row)) * Dm + h * HDm + off;
#if USE_ASYNC_LDS
    async_copy_b128(&qb[g], &Qs[row * HDm + off]);
    async_copy_b128(&kb[g], &Ks[row * HDm + off]);
#else
    *(uint4*)&Qs[row * HDm + off] = *(const uint4*)&qb[g];
    *(uint4*)&Ks[row * HDm + off] = *(const uint4*)&kb[g];
#endif
    Pack8 pv; pv.u = *(const uint4*)&vb[g];   // V needs transpose: VGPR path
#pragma unroll
    for (int t = 0; t < 8; ++t) smVt[(off + t) * BSz + row] = pv.h[t];
  }
  wait_async_zero();
  __syncthreads();

  // ---- scores = Q K^T (16x128 per wave), no 1/sqrt(hd) (faithful) ----
  const v8f vzero = {0.f, 0.f, 0.f, 0.f, 0.f, 0.f, 0.f, 0.f};
  v8f sc[8];
#pragma unroll
  for (int t = 0; t < 8; ++t) sc[t] = vzero;

#pragma unroll
  for (int ks = 0; ks < 2; ++ks) {
    v16bf aq = load_a_frag(Qs, 16 * wv, HDm, 32 * ks);
#pragma unroll
    for (int t = 0; t < 8; ++t) {
      v16bf bk_ = load_b_frag(Ks, 16 * t, HDm, 32 * ks); // row n of K = col of K^T
      sc[t] = wmma_bf16(aq, bk_, sc[t]);
    }
  }

  // ---- softmax over 128 columns, rows resolved by (reg, lane-half) ----
#pragma unroll
  for (int r = 0; r < 8; ++r) {
    float mx = sc[0][r];
#pragma unroll
    for (int t = 1; t < 8; ++t) mx = fmaxf(mx, sc[t][r]);
#pragma unroll
    for (int m = 1; m < 16; m <<= 1) mx = fmaxf(mx, __shfl_xor(mx, m));
    float sum = 0.f;
#pragma unroll
    for (int t = 0; t < 8; ++t) {
      float e = __expf(sc[t][r] - mx);
      sc[t][r] = e;
      sum += e;
    }
#pragma unroll
    for (int m = 1; m < 16; m <<= 1) sum += __shfl_xor(sum, m);
    float inv = 1.0f / sum;
#pragma unroll
    for (int t = 0; t < 8; ++t) sc[t][r] *= inv;
  }

  __syncthreads();   // everyone done reading Q/K fragments -> safe to overlay P
#pragma unroll
  for (int t = 0; t < 8; ++t)
#pragma unroll
    for (int r = 0; r < 8; ++r) {
      int m  = 16 * wv + r + 8 * (lane >> 4);
      int cl = (lane & 15) + 16 * t;
      Ps[m * BSz + cl] = (__bf16)sc[t][r];
    }
  __syncthreads();

  // ---- O = P @ V  (16x64 per wave) ----
  v8f oc[4];
#pragma unroll
  for (int nt = 0; nt < 4; ++nt) oc[nt] = vzero;

#pragma unroll
  for (int ks = 0; ks < 4; ++ks) {
    v16bf ap = load_a_frag(Ps, 16 * wv, BSz, 32 * ks);
#pragma unroll
    for (int nt = 0; nt < 4; ++nt) {
      v16bf bvf = load_b_frag(smVt, 16 * nt, BSz, 32 * ks); // Vt[n][k]
      oc[nt] = wmma_bf16(ap, bvf, oc[nt]);
    }
  }

  // ---- residual add + f32 store ----
#pragma unroll
  for (int nt = 0; nt < 4; ++nt)
#pragma unroll
    for (int r = 0; r < 8; ++r) {
      int m  = 16 * wv + r + 8 * (lane >> 4);
      int cl = 16 * nt + (lane & 15);
      size_t g = ((size_t)(b * Sseq + s0 + m)) * Dm + h * HDm + cl;
      out[g] = oc[nt][r] + x[g];
    }
}

// ---------------------------------------------------------------------------
extern "C" void kernel_launch(void* const* d_in, const int* in_sizes, int n_in,
                              void* d_out, int out_size, void* d_ws,
                              size_t ws_size, hipStream_t stream) {
  (void)in_sizes; (void)n_in; (void)out_size; (void)ws_size;

  const float* x  = (const float*)d_in[0];
  const float* Wq = (const float*)d_in[1];
  const float* bq = (const float*)d_in[2];
  const float* Wk = (const float*)d_in[3];
  const float* bk = (const float*)d_in[4];
  const float* Wv = (const float*)d_in[5];
  const float* bv = (const float*)d_in[6];
  float* out = (float*)d_out;

  char* ws = (char*)d_ws;
  const size_t nTok   = (size_t)Bb * Sseq;     // 16384
  const size_t xElems = nTok * Dm;             // 16,777,216
  const size_t wElems = (size_t)Dm * Dm;       // 1,048,576

  // Workspace layout (bf16): x | Wq | Wk | Wv | Q | K | V   (~140.5 MB)
  __bf16* xb  = (__bf16*)(ws);
  __bf16* wqb = (__bf16*)(ws + xElems * 2);
  __bf16* wkb = (__bf16*)(ws + xElems * 2 + wElems * 2);
  __bf16* wvb = (__bf16*)(ws + xElems * 2 + wElems * 4);
  __bf16* qbuf = (__bf16*)(ws + xElems * 2 + wElems * 6);
  __bf16* kbuf = qbuf + xElems;
  __bf16* vbuf = kbuf + xElems;

  // 0) f32 -> bf16
  f32_to_bf16_kernel<<<(int)(xElems / 4 / 256), 256, 0, stream>>>(x, xb, (int)(xElems / 4));
  f32_to_bf16_kernel<<<(int)(wElems / 4 / 256), 256, 0, stream>>>(Wq, wqb, (int)(wElems / 4));
  f32_to_bf16_kernel<<<(int)(wElems / 4 / 256), 256, 0, stream>>>(Wk, wkb, (int)(wElems / 4));
  f32_to_bf16_kernel<<<(int)(wElems / 4 / 256), 256, 0, stream>>>(Wv, wvb, (int)(wElems / 4));

  // 1) Q/K/V projections: y = x @ W^T + b
  dim3 gGemm(Dm / 128, (unsigned)(nTok / 128));   // (8, 128)
  qkv_gemm_kernel<<<gGemm, 256, 0, stream>>>(xb, wqb, bq, qbuf);
  qkv_gemm_kernel<<<gGemm, 256, 0, stream>>>(xb, wkb, bk, kbuf);
  qkv_gemm_kernel<<<gGemm, 256, 0, stream>>>(xb, wvb, bv, vbuf);

  // 2) bucketed attention + residual
  bucket_attn_kernel<<<Bb * NBkt * Hh, 256, 0, stream>>>(qbuf, kbuf, vbuf, x, out);
}
